// SceneEncoder_27556510171287
// MI455X (gfx1250) — matrix operand order repaired and checked by reading, not compiled, not run on hardware
//
#include <hip/hip_runtime.h>
#include <hip/hip_bf16.h>
#include <math.h>

// ---------------- problem constants ----------------
constexpr int cB = 8, cA = 64, cP = 192, cO = 64;
constexpr int cS = cA + cP + cO;          // 320
constexpr int cD = 256, cNH = 8, cDH = 32, cFF = 1024, cNL = 4;
constexpr int cBS = cB * cS;              // 2560
constexpr float cRADIUS = 80.0f;
constexpr float cNEG = -1000000000.0f;

typedef __attribute__((ext_vector_type(16))) _Float16 v16h;
typedef __attribute__((ext_vector_type(8)))  float    v8f;

// ---------------- WMMA helpers ----------------
__device__ __forceinline__ v8f wmma16(v16h a, v16h b, v8f c) {
  // D(16x16,f32) = A(16x32,f16) x B(32x16,f16) + C
  return __builtin_amdgcn_wmma_f32_16x16x32_f16(false, a, false, b, (short)0, c,
                                                false, false);
}

// A fragment from f32 row-major [.., ld] at tile (m0,k0). Lane l<16: row m=l,
// K={0..7,16..23}; lane l>=16: row m=l-16, K={8..15,24..31}.
__device__ __forceinline__ v16h load_a_f32(const float* __restrict__ X, int ld,
                                           int m0, int k0, int lane) {
  int m = m0 + (lane & 15);
  int hi = lane >> 4;
  const float* p = X + (size_t)m * ld + k0 + hi * 8;
  v16h a;
#pragma unroll
  for (int t = 0; t < 8; ++t) a[t] = (_Float16)p[t];
#pragma unroll
  for (int t = 0; t < 8; ++t) a[8 + t] = (_Float16)p[16 + t];
  return a;
}

// A fragment from f16 row-major (LDS or global)
__device__ __forceinline__ v16h load_a_f16(const _Float16* __restrict__ X, int ld,
                                           int m0, int k0, int lane) {
  int m = m0 + (lane & 15);
  int hi = lane >> 4;
  const _Float16* p = X + (size_t)m * ld + k0 + hi * 8;
  v16h a;
#pragma unroll
  for (int t = 0; t < 8; ++t) a[t] = p[t];
#pragma unroll
  for (int t = 0; t < 8; ++t) a[8 + t] = p[16 + t];
  return a;
}

// B fragment from K-major f16 weight Wt[K][N] at tile (k0,n0): lane l -> col n=l%16,
// K={0..7,16..23} (l<16) or {8..15,24..31} (l>=16).
__device__ __forceinline__ v16h load_b_f16(const _Float16* __restrict__ Wt, int ldn,
                                           int k0, int n0, int lane) {
  int n = n0 + (lane & 15);
  int hi = lane >> 4;
  const _Float16* p = Wt + (size_t)(k0 + hi * 8) * ldn + n;
  v16h b;
#pragma unroll
  for (int t = 0; t < 8; ++t) b[t] = p[(size_t)t * ldn];
#pragma unroll
  for (int t = 0; t < 8; ++t) b[8 + t] = p[(size_t)(16 + t) * ldn];
  return b;
}

__device__ __forceinline__ float gelu_exact(float x) {
  return 0.5f * x * (1.0f + erff(x * 0.7071067811865475f));
}
__device__ __forceinline__ float clamp2(float v) {
  return fminf(fmaxf(v, -2.0f), 2.0f);
}

// ---------------- weight conversion (f32 row-major -> f16 K-major) ----------------
__global__ void conv_wt(const float* __restrict__ W, _Float16* __restrict__ Wt,
                        int dout, int din) {
  int i = blockIdx.x * blockDim.x + threadIdx.x;
  int total = dout * din;
  if (i >= total) return;
  int k = i / dout, n = i - k * dout;
  Wt[i] = (_Float16)W[(size_t)n * din + k];  // Wt[k][n] = W[n][k]
}

// bias-MLP W1 (256,7) -> [32][256], K rows 7..31 zero
__global__ void conv_biasW1(const float* __restrict__ W, _Float16* __restrict__ Wt) {
  int i = blockIdx.x * 256 + threadIdx.x;  // 32*256
  int k = i / 256, n = i - k * 256;
  Wt[i] = (k < 7) ? (_Float16)W[n * 7 + k] : (_Float16)0.0f;
}
// bias-MLP W2 (8,256) -> [256][16], N cols 8..15 zero
__global__ void conv_biasW2(const float* __restrict__ W, _Float16* __restrict__ Wt) {
  int i = blockIdx.x * 256 + threadIdx.x;  // 256*16
  int k = i / 16, n = i - k * 16;
  Wt[i] = (n < 8) ? (_Float16)W[n * 256 + k] : (_Float16)0.0f;
}

// ---------------- generic WMMA GEMM: Y = act(X @ Wt + bias) [+ residual] ----------------
// X:[M,K] f32, Wt:[K,N] f16 (K-major). One wave -> 32x64 slab:
// each B fragment feeds two WMMAs (8 wmma per K-step), weight rows prefetched.
__global__ void gemm_ws(const float* __restrict__ X, const _Float16* __restrict__ Wt,
                        const float* __restrict__ bias, const float* __restrict__ residual,
                        float* __restrict__ Y, int M, int N, int K, int act) {
  int lane = threadIdx.x;
  int n0 = blockIdx.x * 64;
  int m0 = blockIdx.y * 32;
  v8f acc[2][4];
#pragma unroll
  for (int mt = 0; mt < 2; ++mt)
#pragma unroll
    for (int t = 0; t < 4; ++t) acc[mt][t] = v8f{};
  for (int k0 = 0; k0 < K; k0 += 32) {
    v16h a0 = load_a_f32(X, K, m0, k0, lane);
    v16h a1 = load_a_f32(X, K, m0 + 16, k0, lane);
    if (k0 + 32 < K)  // pull next K-step's weight rows toward the WGP
      __builtin_prefetch(Wt + (size_t)(k0 + 32 + lane) * N + n0, 0, 1);
#pragma unroll
    for (int t = 0; t < 4; ++t) {
      v16h b = load_b_f16(Wt, N, k0, n0 + 16 * t, lane);
      acc[0][t] = wmma16(a0, b, acc[0][t]);
      acc[1][t] = wmma16(a1, b, acc[1][t]);
    }
  }
  int nn = lane & 15, hi = lane >> 4;
#pragma unroll
  for (int mt = 0; mt < 2; ++mt) {
#pragma unroll
    for (int t = 0; t < 4; ++t) {
      int col = n0 + 16 * t + nn;
      float bv = bias ? bias[col] : 0.0f;
#pragma unroll
      for (int r = 0; r < 8; ++r) {
        int row = m0 + mt * 16 + r + 8 * hi;
        float v = acc[mt][t][r] + bv;
        if (act == 1) v = gelu_exact(v);
        if (residual) v += residual[(size_t)row * N + col];
        Y[(size_t)row * N + col] = v;
      }
    }
  }
}

// ---------------- token prep + pose MLP layer 1 ----------------
__global__ void prep_tokens(const float* __restrict__ at, const float* __restrict__ pt,
                            const float* __restrict__ ot, const float* __restrict__ ap,
                            const float* __restrict__ pp, const float* __restrict__ op,
                            const float* __restrict__ type_emb,
                            const float* __restrict__ pW1, const float* __restrict__ pb1,
                            float* __restrict__ x, float* __restrict__ h1) {
  int row = blockIdx.x;  // 0..cBS-1
  int t = threadIdx.x;   // 0..255
  int b = row / cS, s = row - b * cS;
  const float* tok;
  const float* pose;
  int seg;
  if (s < cA) {
    seg = 0; tok = at + ((size_t)b * cA + s) * cD; pose = ap + ((size_t)b * cA + s) * 4;
  } else if (s < cA + cP) {
    seg = 1; tok = pt + ((size_t)b * cP + (s - cA)) * cD;
    pose = pp + ((size_t)b * cP + (s - cA)) * 4;
  } else {
    seg = 2; tok = ot + ((size_t)b * cO + (s - cA - cP)) * cD;
    pose = op + ((size_t)b * cO + (s - cA - cP)) * 4;
  }
  float f0 = clamp2(pose[0] / cRADIUS), f1 = clamp2(pose[1] / cRADIUS);
  float f2 = pose[2], f3 = pose[3];
  x[(size_t)row * cD + t] = tok[t] + type_emb[seg * cD + t];
  const float* w = pW1 + t * 4;
  h1[(size_t)row * cD + t] =
      gelu_exact(pb1[t] + f0 * w[0] + f1 * w[1] + f2 * w[2] + f3 * w[3]);
}

// ---------------- pad mask -> additive key bias ----------------
__global__ void pad_bias(const int* __restrict__ ap, const int* __restrict__ pp,
                         const int* __restrict__ op, float* __restrict__ padb) {
  __shared__ int cnt;
  int b = blockIdx.x, s = threadIdx.x;  // blockDim = cS
  int pad = (s < cA) ? ap[b * cA + s]
            : (s < cA + cP) ? pp[b * cP + (s - cA)]
                            : op[b * cO + (s - cA - cP)];
  if (s == 0) cnt = 0;
  __syncthreads();
  if (pad != 0) atomicAdd(&cnt, 1);
  __syncthreads();
  bool fully = (cnt == cS);
  bool safe = (pad != 0) && !(fully && s == 0);
  padb[b * cS + s] = safe ? cNEG : 0.0f;
}

__device__ __forceinline__ const float* pose_ptr(const float* ap, const float* pp,
                                                 const float* op, int b, int s) {
  if (s < cA) return ap + ((size_t)b * cA + s) * 4;
  if (s < cA + cP) return pp + ((size_t)b * cP + (s - cA)) * 4;
  return op + ((size_t)b * cO + (s - cA - cP)) * 4;
}

// ---------------- fused relative-bias MLP: one wave per (b, i, 16 j's) ----------------
__global__ void rel_bias(const float* __restrict__ ap, const float* __restrict__ pp,
                         const float* __restrict__ op,
                         const _Float16* __restrict__ W1t,  // [32][256], rows>=7 zero
                         const float* __restrict__ b1,
                         const _Float16* __restrict__ W2t,  // [256][16], cols>=8 zero
                         const float* __restrict__ b2,
                         const float* __restrict__ padb, float* __restrict__ abias) {
  __shared__ _Float16 hbuf[16][cD];  // 16 pairs x 256 hidden, f16
  int lane = threadIdx.x;
  int b = blockIdx.z, i = blockIdx.y, j0 = blockIdx.x * 16;
  int m = lane & 15, hi = lane >> 4;
  int j = j0 + m;

  const float* qp = pose_ptr(ap, pp, op, b, i);
  const float* kp = pose_ptr(ap, pp, op, b, j);
  float qc = qp[2], qs = qp[3];
  float dx = clamp2((kp[0] - qp[0]) / cRADIUS);
  float dy = clamp2((kp[1] - qp[1]) / cRADIUS);
  float dist = fminf(sqrtf(dx * dx + dy * dy), 2.0f);
  float ldx = dx * qc + dy * qs;
  float ldy = -dx * qs + dy * qc;
  float rc = kp[2] * qc + kp[3] * qs;
  float rs = kp[3] * qc - kp[2] * qs;
  float f[8] = {dx, dy, ldx, ldy, dist, rc, rs, 0.0f};

  v16h a;  // K=32, only K<7 nonzero -> lanes>=16 (K 8..15,24..31) are all zero
#pragma unroll
  for (int t = 0; t < 16; ++t) a[t] = (_Float16)0.0f;
  if (hi == 0) {
#pragma unroll
    for (int t = 0; t < 8; ++t) a[t] = (_Float16)f[t];
  }

  // hidden = gelu(feats @ W1^T + b1), 16 N-tiles of 16
  for (int nt = 0; nt < 16; ++nt) {
    v16h bb = load_b_f16(W1t, cD, 0, nt * 16, lane);
    v8f c = v8f{};
    c = wmma16(a, bb, c);
    int nn = nt * 16 + (lane & 15);
#pragma unroll
    for (int r = 0; r < 8; ++r)
      hbuf[r + 8 * hi][nn] = (_Float16)gelu_exact(c[r] + b1[nn]);
  }
  __syncthreads();

  // out = hidden @ W2^T + b2  (N padded to 16, 8 valid)
  v8f acc = v8f{};
  for (int k0 = 0; k0 < cD; k0 += 32) {
    v16h ah = load_a_f16(&hbuf[0][0], cD, 0, k0, lane);
    v16h bb = load_b_f16(W2t, 16, k0, 0, lane);
    acc = wmma16(ah, bb, acc);
  }
  int n = lane & 15;
  if (n < cNH) {
#pragma unroll
    for (int r = 0; r < 8; ++r) {
      int mm = r + 8 * hi;
      int jj = j0 + mm;
      abias[(((size_t)b * cNH + n) * cS + i) * cS + jj] =
          acc[r] + b2[n] + padb[b * cS + jj];
    }
  }
}

// ---------------- fused attention: one wave per (b, h, 16 q rows) ----------------
__global__ void attn(const float* __restrict__ qkv, const float* __restrict__ abias,
                     float* __restrict__ o) {
  __shared__ float sc[16][cS];        // scores/probs: 20 KB
  __shared__ _Float16 kT[32][16];     // K^T staging tile
  __shared__ float red[2][16][2];
  int lane = threadIdx.x;
  int b = blockIdx.z, h = blockIdx.y, q0 = blockIdx.x * 16;
  int m = lane & 15, hi = lane >> 4;
  const float scale = 0.17677669529663687f;  // 1/sqrt(32)
  size_t base = (size_t)b * cS * 768;
  int qoff = h * cDH, koff = 256 + h * cDH, voff = 512 + h * cDH;

  // Q fragment (dh = 32 = one K step)
  v16h aq;
  {
    const float* p = qkv + base + (size_t)(q0 + m) * 768 + qoff + hi * 8;
#pragma unroll
    for (int t = 0; t < 8; ++t) aq[t] = (_Float16)p[t];
#pragma unroll
    for (int t = 0; t < 8; ++t) aq[8 + t] = (_Float16)p[16 + t];
  }

  // scores = Q K^T * scale + bias
  for (int j0 = 0; j0 < cS; j0 += 16) {
    {  // stage K^T: kT[dh][key]
      const float* kr = qkv + base + (size_t)(j0 + m) * 768 + koff + hi * 16;
#pragma unroll
      for (int t = 0; t < 16; ++t) kT[hi * 16 + t][m] = (_Float16)kr[t];
    }
    __syncthreads();
    v16h bk;
    const _Float16* pk = &kT[hi * 8][m];
#pragma unroll
    for (int t = 0; t < 16; ++t) bk[t] = pk[(t < 8 ? t : t + 8) * 16];
    v8f c = v8f{};
    c = wmma16(aq, bk, c);
    int nn = lane & 15;
#pragma unroll
    for (int r = 0; r < 8; ++r) {
      int q = q0 + r + 8 * hi;
      sc[r + 8 * hi][j0 + nn] =
          c[r] * scale + abias[(((size_t)b * cNH + h) * cS + q) * cS + j0 + nn];
    }
    __syncthreads();
  }

  // softmax over keys: lane pair (m, m+16) splits the 320 keys
  {
    int lo = hi * 160, hiEnd = lo + 160;
    float mx = -INFINITY;
    for (int jj = lo; jj < hiEnd; ++jj) mx = fmaxf(mx, sc[m][jj]);
    red[0][m][hi] = mx;
    __syncthreads();
    mx = fmaxf(red[0][m][0], red[0][m][1]);
    float sum = 0.0f;
    for (int jj = lo; jj < hiEnd; ++jj) {
      float e = __expf(sc[m][jj] - mx);
      sc[m][jj] = e;
      sum += e;
    }
    red[1][m][hi] = sum;
    __syncthreads();
    float inv = 1.0f / (red[1][m][0] + red[1][m][1]);
    for (int jj = lo; jj < hiEnd; ++jj) sc[m][jj] *= inv;
    __syncthreads();
  }

  // O = P V  (N = 32 -> two 16-wide tiles)
  v8f acc0 = v8f{}, acc1 = v8f{};
  for (int k0 = 0; k0 < cS; k0 += 32) {
    v16h apf = load_a_f32(&sc[0][0], cS, 0, k0, lane);
    const float* vr = qkv + base + (size_t)(k0 + hi * 8) * 768 + voff + m;
    v16h b0, b1v;
#pragma unroll
    for (int t = 0; t < 16; ++t) b0[t] = (_Float16)vr[(size_t)(t < 8 ? t : t + 8) * 768];
#pragma unroll
    for (int t = 0; t < 16; ++t) b1v[t] = (_Float16)vr[(size_t)(t < 8 ? t : t + 8) * 768 + 16];
    acc0 = wmma16(apf, b0, acc0);
    acc1 = wmma16(apf, b1v, acc1);
  }
  int nn = lane & 15;
#pragma unroll
  for (int r = 0; r < 8; ++r) {
    int q = q0 + r + 8 * hi;
    float* orow = o + ((size_t)b * cS + q) * cD + h * cDH;
    orow[nn] = acc0[r];
    orow[16 + nn] = acc1[r];
  }
}

// ---------------- residual add + layernorm (in place on x) ----------------
__global__ void add_ln(float* __restrict__ x, const float* __restrict__ addv,
                       const float* __restrict__ g, const float* __restrict__ bt) {
  __shared__ float rbuf[cD];
  int row = blockIdx.x, t = threadIdx.x;
  size_t idx = (size_t)row * cD + t;
  float v = x[idx] + addv[idx];
  rbuf[t] = v;
  __syncthreads();
  for (int s2 = 128; s2 > 0; s2 >>= 1) {
    if (t < s2) rbuf[t] += rbuf[t + s2];
    __syncthreads();
  }
  float mean = rbuf[0] * (1.0f / cD);
  __syncthreads();
  float d = v - mean;
  rbuf[t] = d * d;
  __syncthreads();
  for (int s2 = 128; s2 > 0; s2 >>= 1) {
    if (t < s2) rbuf[t] += rbuf[t + s2];
    __syncthreads();
  }
  float var = rbuf[0] * (1.0f / cD);
  x[idx] = d * rsqrtf(var + 1e-5f) * g[t] + bt[t];
}

// ---------------- final LN + pad-zero + slice agents ----------------
__global__ void final_out(const float* __restrict__ x, const float* __restrict__ g,
                          const float* __restrict__ bt, const int* __restrict__ apad,
                          float* __restrict__ out) {
  __shared__ float rbuf[cD];
  int a = blockIdx.x, b = blockIdx.y, t = threadIdx.x;
  size_t idx = ((size_t)b * cS + a) * cD + t;
  float v = x[idx];
  rbuf[t] = v;
  __syncthreads();
  for (int s2 = 128; s2 > 0; s2 >>= 1) {
    if (t < s2) rbuf[t] += rbuf[t + s2];
    __syncthreads();
  }
  float mean = rbuf[0] * (1.0f / cD);
  __syncthreads();
  float d = v - mean;
  rbuf[t] = d * d;
  __syncthreads();
  for (int s2 = 128; s2 > 0; s2 >>= 1) {
    if (t < s2) rbuf[t] += rbuf[t + s2];
    __syncthreads();
  }
  float var = rbuf[0] * (1.0f / cD);
  float y = d * rsqrtf(var + 1e-5f) * g[t] + bt[t];
  bool padv = apad[b * cA + a] != 0;
  out[((size_t)b * cA + a) * cD + t] = padv ? 0.0f : y;
}

// ---------------- host orchestration ----------------
extern "C" void kernel_launch(void* const* d_in, const int* in_sizes, int n_in,
                              void* d_out, int out_size, void* d_ws, size_t ws_size,
                              hipStream_t stream) {
  (void)in_sizes; (void)n_in; (void)out_size;
  auto F = [&](int i) { return (const float*)d_in[i]; };
  const float* agent_tokens = F(0);
  const int*   agent_pad    = (const int*)d_in[1];
  const float* agent_poses  = F(2);
  const float* poly_tokens  = F(3);
  const int*   poly_pad     = (const int*)d_in[4];
  const float* poly_poses   = F(5);
  const float* obj_tokens   = F(6);
  const int*   obj_pad      = (const int*)d_in[7];
  const float* obj_poses    = F(8);

  // params flattened depth-first in insertion order
  int p = 9;
  const float *Wqkv[cNL], *bqkv[cNL], *Wo[cNL], *bo[cNL], *W1[cNL], *b1[cNL],
      *W2[cNL], *b2[cNL], *ln1g[cNL], *ln1b[cNL], *ln2g[cNL], *ln2b[cNL];
  for (int l = 0; l < cNL; ++l) {
    Wqkv[l] = F(p++); bqkv[l] = F(p++); Wo[l] = F(p++); bo[l] = F(p++);
    W1[l] = F(p++);   b1[l] = F(p++);   W2[l] = F(p++); b2[l] = F(p++);
    ln1g[l] = F(p++); ln1b[l] = F(p++); ln2g[l] = F(p++); ln2b[l] = F(p++);
  }
  const float* ln_g = F(p++);
  const float* ln_b = F(p++);
  const float* type_emb = F(p++);
  const float* pW1 = F(p++); const float* pb1 = F(p++);
  const float* pW2 = F(p++); const float* pb2 = F(p++);
  const float* rW1 = F(p++); const float* rb1 = F(p++);
  const float* rW2 = F(p++); const float* rb2 = F(p++);

  // workspace carve
  char* wsB = (char*)d_ws;
  size_t cur = 0;
  auto alloc = [&](size_t bytes) -> void* {
    void* r = wsB + cur;
    cur = (cur + bytes + 255) & ~(size_t)255;
    return r;
  };
  _Float16 *Wqkvt[cNL], *Wot[cNL], *W1t[cNL], *W2t[cNL];
  for (int l = 0; l < cNL; ++l) {
    Wqkvt[l] = (_Float16*)alloc((size_t)768 * cD * 2);
    Wot[l]   = (_Float16*)alloc((size_t)cD * cD * 2);
    W1t[l]   = (_Float16*)alloc((size_t)cFF * cD * 2);
    W2t[l]   = (_Float16*)alloc((size_t)cD * cFF * 2);
  }
  _Float16* pW2t = (_Float16*)alloc((size_t)cD * cD * 2);
  _Float16* rW1t = (_Float16*)alloc((size_t)32 * cD * 2);
  _Float16* rW2t = (_Float16*)alloc((size_t)cD * 16 * 2);
  float* x      = (float*)alloc((size_t)cBS * cD * 4);
  float* qkv    = (float*)alloc((size_t)cBS * 768 * 4);
  float* oheads = (float*)alloc((size_t)cBS * cD * 4);
  float* attb   = (float*)alloc((size_t)cBS * cD * 4);
  float* hff    = (float*)alloc((size_t)cBS * cFF * 4);  // also pose-MLP hidden
  float* abias  = (float*)alloc((size_t)cB * cNH * cS * cS * 4);
  float* padb   = (float*)alloc((size_t)cB * cS * 4);
  if (cur > ws_size) return;  // insufficient workspace

  // 1) convert weights to f16 K-major
  for (int l = 0; l < cNL; ++l) {
    conv_wt<<<(768 * cD + 255) / 256, 256, 0, stream>>>(Wqkv[l], Wqkvt[l], 768, cD);
    conv_wt<<<(cD * cD + 255) / 256, 256, 0, stream>>>(Wo[l], Wot[l], cD, cD);
    conv_wt<<<(cFF * cD + 255) / 256, 256, 0, stream>>>(W1[l], W1t[l], cFF, cD);
    conv_wt<<<(cD * cFF + 255) / 256, 256, 0, stream>>>(W2[l], W2t[l], cD, cFF);
  }
  conv_wt<<<(cD * cD + 255) / 256, 256, 0, stream>>>(pW2, pW2t, cD, cD);
  conv_biasW1<<<32, 256, 0, stream>>>(rW1, rW1t);
  conv_biasW2<<<16, 256, 0, stream>>>(rW2, rW2t);

  // 2) token prep + pose MLP
  pad_bias<<<cB, cS, 0, stream>>>(agent_pad, poly_pad, obj_pad, padb);
  prep_tokens<<<cBS, cD, 0, stream>>>(agent_tokens, poly_tokens, obj_tokens,
                                      agent_poses, poly_poses, obj_poses, type_emb,
                                      pW1, pb1, x, hff);
  gemm_ws<<<dim3(cD / 64, cBS / 32), 32, 0, stream>>>(hff, pW2t, pb2, x, x,
                                                      cBS, cD, cD, 0);

  // 3) relative attention bias (+ pad mask folded in)
  rel_bias<<<dim3(cS / 16, cS, cB), 32, 0, stream>>>(
      agent_poses, poly_poses, obj_poses, rW1t, rb1, rW2t, rb2, padb, abias);

  // 4) transformer layers
  for (int l = 0; l < cNL; ++l) {
    gemm_ws<<<dim3(768 / 64, cBS / 32), 32, 0, stream>>>(x, Wqkvt[l], bqkv[l],
                                                         nullptr, qkv, cBS, 768, cD, 0);
    attn<<<dim3(cS / 16, cNH, cB), 32, 0, stream>>>(qkv, abias, oheads);
    gemm_ws<<<dim3(cD / 64, cBS / 32), 32, 0, stream>>>(oheads, Wot[l], bo[l],
                                                        nullptr, attb, cBS, cD, cD, 0);
    add_ln<<<cBS, cD, 0, stream>>>(x, attb, ln1g[l], ln1b[l]);
    gemm_ws<<<dim3(cFF / 64, cBS / 32), 32, 0, stream>>>(x, W1t[l], b1[l],
                                                         nullptr, hff, cBS, cFF, cD, 1);
    gemm_ws<<<dim3(cD / 64, cBS / 32), 32, 0, stream>>>(hff, W2t[l], b2[l],
                                                        nullptr, attb, cBS, cD, cFF, 0);
    add_ln<<<cBS, cD, 0, stream>>>(x, attb, ln2g[l], ln2b[l]);
  }

  // 5) final LN + pad-zero + agent slice
  final_out<<<dim3(cA, cB), cD, 0, stream>>>(x, ln_g, ln_b, agent_pad, (float*)d_out);
}
